// MambaBlock_83717502533805
// MI455X (gfx1250) — matrix operand order repaired and checked by reading, not compiled, not run on hardware
//
#include <hip/hip_runtime.h>
#include <hip/hip_bf16.h>

// ---------------------------------------------------------------------------
// Types for CDNA5 WMMA (wave32, v_wmma_f32_16x16x32_bf16)
// ---------------------------------------------------------------------------
typedef __bf16        v16bf __attribute__((ext_vector_type(16)));
typedef float         v8f   __attribute__((ext_vector_type(8)));
typedef unsigned int  v8u   __attribute__((ext_vector_type(8)));

struct bf16x16_bits { uint4 lo, hi; }; // 32 bytes, bit-castable to v16bf

static __device__ __forceinline__ unsigned short f2bf(float f) {
    union { float f; unsigned u; } c{f};
    unsigned r = c.u + 0x7fffu + ((c.u >> 16) & 1u); // round-to-nearest-even
    return (unsigned short)(r >> 16);
}
static __device__ __forceinline__ float sigmoidf_(float v) { return 1.0f / (1.0f + __expf(-v)); }
static __device__ __forceinline__ float siluf_(float v)    { return v * sigmoidf_(v); }
static __device__ __forceinline__ float softplusf_(float v){ return v > 20.0f ? v : log1pf(__expf(v)); }
static __device__ __forceinline__ float gelu_tanh_(float v){
    float c = 0.7978845608028654f * (v + 0.044715f * v * v * v);
    return 0.5f * v * (1.0f + tanhf(c));
}

static __device__ __forceinline__ v16bf load_a_frag(const unsigned short* p) {
    bf16x16_bits ab;
    ab.lo = *(const uint4*)p;         // K = kh*16 + 0..7
    ab.hi = *(const uint4*)(p + 8);   // K = kh*16 + 8..15
    return __builtin_bit_cast(v16bf, ab);
}
static __device__ __forceinline__ v16bf load_b_frag(const unsigned short* p) {
    return __builtin_bit_cast(v16bf, *(const v8u*)p); // K = kh*16 + 0..15
}
static __device__ __forceinline__ v8f wmma_bf16(v16bf a, v16bf b, v8f c) {
    return __builtin_amdgcn_wmma_f32_16x16x32_bf16(false, a, false, b, (short)0, c, false, false);
}

// ---------------------------------------------------------------------------
// Problem constants
// ---------------------------------------------------------------------------
static constexpr int Bb = 2, Ls = 4096, Dd = 1024, Di = 1024, Ns = 16, Rr = 64, Hh = 4096;
static constexpr int Mrows = Bb * Ls; // 8192

// ---------------------------------------------------------------------------
// fp32 -> bf16 conversion (with zero padding up to ntot)
// ---------------------------------------------------------------------------
__global__ void k_f32_to_bf16(const float* __restrict__ in, unsigned short* __restrict__ out,
                              long long n, long long ntot) {
    long long i = (long long)blockIdx.x * blockDim.x + threadIdx.x;
    if (i < ntot) out[i] = (i < n) ? f2bf(in[i]) : (unsigned short)0;
}

// ---------------------------------------------------------------------------
// gather (x[:, path, :]) + LayerNorm -> bf16    (one block per row, 256 thr)
// ---------------------------------------------------------------------------
__global__ __launch_bounds__(256) void k_gather_ln(const float* __restrict__ x,
                                                   const int* __restrict__ path,
                                                   unsigned short* __restrict__ out) {
    __shared__ float s1[256], s2[256];
    const int row = blockIdx.x;           // b * L + l (permuted index)
    const int b = row / Ls, l = row % Ls;
    const int src = path[l];
    const float* p = x + ((size_t)b * Ls + src) * Dd;
    float v[4], sum = 0.f, sq = 0.f;
#pragma unroll
    for (int i = 0; i < 4; ++i) { v[i] = p[threadIdx.x + i * 256]; sum += v[i]; sq += v[i] * v[i]; }
    s1[threadIdx.x] = sum; s2[threadIdx.x] = sq; __syncthreads();
    for (int st = 128; st > 0; st >>= 1) {
        if (threadIdx.x < st) { s1[threadIdx.x] += s1[threadIdx.x + st]; s2[threadIdx.x] += s2[threadIdx.x + st]; }
        __syncthreads();
    }
    const float mean = s1[0] * (1.0f / Dd);
    const float var  = s2[0] * (1.0f / Dd) - mean * mean;
    const float inv  = rsqrtf(var + 1e-6f);
    unsigned short* o = out + (size_t)row * Dd;
#pragma unroll
    for (int i = 0; i < 4; ++i) o[threadIdx.x + i * 256] = f2bf((v[i] - mean) * inv);
}

// ---------------------------------------------------------------------------
// WMMA GEMM:  D[M,N] = A[M,K] * W[N,K]^T  (bf16 in, fp32 accumulate)
// Block = 256 threads = 8 waves (4 M x 2 N); block tile 128M x 128N;
// wave tile 32M x 64N = 8 WMMA accumulators -> 1.5 b128 loads per WMMA.
// Fragment k order: contiguous k = 16*(lane/16) + s in A and B (16-bit A 16x32, B 32x16).
// EPI: 0 = f32 store | 1 = bias+softplus f32 | 2 = bias+gelu bf16 | 3 = bias+residual f32
// ---------------------------------------------------------------------------
template <int EPI>
__global__ __launch_bounds__(256) void gemm_bf16_wmma(
    const unsigned short* __restrict__ A, const unsigned short* __restrict__ W,
    float* __restrict__ Cf, unsigned short* __restrict__ Cb,
    const float* __restrict__ bias, const float* __restrict__ res,
    int M, int N, int K) {
    const int tid  = threadIdx.x;
    const int wave = tid >> 5, lane = tid & 31;
    const int mw = wave >> 1;       // 0..3
    const int nw = wave & 1;        // 0..1
    const int row0 = blockIdx.y * 128 + mw * 32;
    const int col0 = blockIdx.x * 128 + nw * 64;
    const int mm = lane & 15, kh = lane >> 4;

    const unsigned short* pa0 = A + (size_t)(row0 + mm) * K + kh * 16;  // M rows 0..15
    const unsigned short* pa1 = pa0 + (size_t)16 * K;                   // M rows 16..31
    const unsigned short* pb  = W + (size_t)(col0 + mm) * K + kh * 16;  // N cols t*16..

    v8f acc[2][4] = {};
    for (int kk = 0; kk < K; kk += 32) {
        const v16bf a0 = load_a_frag(pa0 + kk);
        const v16bf a1 = load_a_frag(pa1 + kk);
        __builtin_prefetch(pa0 + kk + 256, 0, 1);
        __builtin_prefetch(pa1 + kk + 256, 0, 1);
        __builtin_prefetch(pb  + kk + 256, 0, 1);
#pragma unroll
        for (int t = 0; t < 4; ++t) {
            const v16bf b = load_b_frag(pb + (size_t)t * 16 * K + kk);
            acc[0][t] = wmma_bf16(a0, b, acc[0][t]);
            acc[1][t] = wmma_bf16(a1, b, acc[1][t]);
        }
    }

#pragma unroll
    for (int mi = 0; mi < 2; ++mi) {
#pragma unroll
        for (int t = 0; t < 4; ++t) {
            const int c = col0 + t * 16 + mm;
#pragma unroll
            for (int i = 0; i < 8; ++i) {
                const int r = row0 + mi * 16 + kh * 8 + i;
                const size_t o = (size_t)r * N + c;
                float v = acc[mi][t][i];
                if (EPI == 0) {
                    Cf[o] = v;
                } else if (EPI == 1) {
                    Cf[o] = softplusf_(v + bias[c]);
                } else if (EPI == 2) {
                    Cb[o] = f2bf(gelu_tanh_(v + bias[c]));
                } else {
                    Cf[o] = res[o] + v + bias[c];
                }
            }
        }
    }
}

// ---------------------------------------------------------------------------
// depthwise causal conv (D_CONV=4) + bias + SiLU; reads xi = xz[..., :Di]
// ---------------------------------------------------------------------------
__global__ void k_conv_silu(const float* __restrict__ xz, const float* __restrict__ conv_w,
                            const float* __restrict__ conv_b, float* __restrict__ xcf,
                            unsigned short* __restrict__ xcb) {
    long long i = (long long)blockIdx.x * blockDim.x + threadIdx.x;
    if (i >= (long long)Mrows * Di) return;
    const int d = (int)(i % Di);
    const int l = (int)((i / Di) % Ls);
    const int b = (int)(i / ((long long)Di * Ls));
    float acc = conv_b[d];
#pragma unroll
    for (int k = 0; k < 4; ++k) {
        const int lk = l - 3 + k;
        if (lk >= 0) acc += conv_w[d * 4 + k] * xz[((size_t)b * Ls + lk) * (2 * Di) + d];
    }
    acc = siluf_(acc);
    xcf[i] = acc;
    xcb[i] = f2bf(acc);
}

// dt_r = proj[:, :64] (ld 128) -> bf16
__global__ void k_slice_dtr(const float* __restrict__ proj, unsigned short* __restrict__ dtr) {
    long long i = (long long)blockIdx.x * blockDim.x + threadIdx.x;
    if (i >= (long long)Mrows * Rr) return;
    const long long m = i / Rr, j = i % Rr;
    dtr[i] = f2bf(proj[m * 128 + j]);
}

// ---------------------------------------------------------------------------
// selective scan: one thread per channel d, sequential over L, LDS-broadcast
// B_t/C_t; fuses D-skip and SiLU(z) gating; emits bf16 for out_proj GEMM.
// ---------------------------------------------------------------------------
__global__ __launch_bounds__(256) void k_scan(const float* __restrict__ proj,
                                              const float* __restrict__ dt,
                                              const float* __restrict__ xcf,
                                              const float* __restrict__ xz,
                                              const float* __restrict__ A_log,
                                              const float* __restrict__ Dp,
                                              unsigned short* __restrict__ ybf) {
    __shared__ float sB[16], sC[16];
    const int b = blockIdx.y;
    const int d = blockIdx.x * 256 + threadIdx.x;
    float a[16], h[16];
#pragma unroll
    for (int n = 0; n < 16; ++n) { a[n] = -__expf(A_log[d * 16 + n]); h[n] = 0.f; }
    const float dskip = Dp[d];
    for (int l = 0; l < Ls; ++l) {
        const size_t row = (size_t)b * Ls + l;
        if (threadIdx.x < 16)       sB[threadIdx.x]      = proj[row * 128 + 64 + threadIdx.x];
        else if (threadIdx.x < 32)  sC[threadIdx.x - 16] = proj[row * 128 + 80 + (threadIdx.x - 16)];
        __syncthreads();
        const float dtv = dt[row * Di + d];
        const float xv  = xcf[row * Di + d];
        const float zv  = xz[row * (2 * Di) + Di + d];
        const float bx  = dtv * xv;
        float y = 0.f;
#pragma unroll
        for (int n = 0; n < 16; ++n) {
            h[n] = __expf(dtv * a[n]) * h[n] + bx * sB[n];
            y += h[n] * sC[n];
        }
        y = (y + dskip * xv) * siluf_(zv);
        ybf[row * Di + d] = f2bf(y);
        __syncthreads();
    }
}

// ---------------------------------------------------------------------------
// residual scatter (x + mo[:, path_rev, :]) -> d_out, plus LayerNorm -> bf16
// ---------------------------------------------------------------------------
__global__ __launch_bounds__(256) void k_resid_ln(const float* __restrict__ x,
                                                  const float* __restrict__ mo,
                                                  const int* __restrict__ path_rev,
                                                  float* __restrict__ out,
                                                  unsigned short* __restrict__ ln2) {
    __shared__ float s1[256], s2[256];
    const int row = blockIdx.x;
    const int b = row / Ls, lo = row % Ls;
    const int lp = path_rev[lo];
    const float* px = x  + (size_t)row * Dd;
    const float* pm = mo + ((size_t)b * Ls + lp) * Dd;
    float v[4], sum = 0.f, sq = 0.f;
#pragma unroll
    for (int i = 0; i < 4; ++i) {
        v[i] = px[threadIdx.x + i * 256] + pm[threadIdx.x + i * 256];
        sum += v[i]; sq += v[i] * v[i];
    }
    s1[threadIdx.x] = sum; s2[threadIdx.x] = sq; __syncthreads();
    for (int st = 128; st > 0; st >>= 1) {
        if (threadIdx.x < st) { s1[threadIdx.x] += s1[threadIdx.x + st]; s2[threadIdx.x] += s2[threadIdx.x + st]; }
        __syncthreads();
    }
    const float mean = s1[0] * (1.0f / Dd);
    const float var  = s2[0] * (1.0f / Dd) - mean * mean;
    const float inv  = rsqrtf(var + 1e-6f);
    float* po = out + (size_t)row * Dd;
    unsigned short* pl = ln2 + (size_t)row * Dd;
#pragma unroll
    for (int i = 0; i < 4; ++i) {
        po[threadIdx.x + i * 256] = v[i];
        pl[threadIdx.x + i * 256] = f2bf((v[i] - mean) * inv);
    }
}

// ---------------------------------------------------------------------------
extern "C" void kernel_launch(void* const* d_in, const int* in_sizes, int n_in,
                              void* d_out, int out_size, void* d_ws, size_t ws_size,
                              hipStream_t stream) {
    const float* x         = (const float*)d_in[0];
    const int*   path      = (const int*)d_in[1];
    const int*   path_rev  = (const int*)d_in[2];
    const float* in_proj_w = (const float*)d_in[3];
    const float* conv_w    = (const float*)d_in[4];
    const float* conv_b    = (const float*)d_in[5];
    const float* x_proj_w  = (const float*)d_in[6];
    const float* dt_proj_w = (const float*)d_in[7];
    const float* dt_proj_b = (const float*)d_in[8];
    const float* A_log     = (const float*)d_in[9];
    const float* D_param   = (const float*)d_in[10];
    const float* out_proj_w= (const float*)d_in[11];
    const float* fc1_w     = (const float*)d_in[12];
    const float* fc1_b     = (const float*)d_in[13];
    const float* fc2_w     = (const float*)d_in[14];
    const float* fc2_b     = (const float*)d_in[15];
    float* out = (float*)d_out;

    char* base = (char*)d_ws;
    size_t off = 0;
    auto alloc = [&](size_t bytes) -> char* {
        char* p = base + off;
        off = (off + bytes + 255) & ~(size_t)255;
        return p;
    };
    unsigned short* w_in  = (unsigned short*)alloc((size_t)2 * Di * Dd * 2);
    unsigned short* w_xp  = (unsigned short*)alloc((size_t)128 * Di * 2);   // padded 96->128 rows
    unsigned short* w_dtp = (unsigned short*)alloc((size_t)Di * Rr * 2);
    unsigned short* w_op  = (unsigned short*)alloc((size_t)Dd * Di * 2);
    unsigned short* w_f1  = (unsigned short*)alloc((size_t)Hh * Dd * 2);
    unsigned short* w_f2  = (unsigned short*)alloc((size_t)Dd * Hh * 2);
    unsigned short* ln1   = (unsigned short*)alloc((size_t)Mrows * Dd * 2); // reused as ln2
    float*          xz    = (float*)alloc((size_t)Mrows * 2 * Di * 4);      // reused as h1 (bf16)
    float*          xcf   = (float*)alloc((size_t)Mrows * Di * 4);
    unsigned short* xcb   = (unsigned short*)alloc((size_t)Mrows * Di * 2);
    float*          proj  = (float*)alloc((size_t)Mrows * 128 * 4);
    unsigned short* dtr   = (unsigned short*)alloc((size_t)Mrows * Rr * 2);
    float*          dtf   = (float*)alloc((size_t)Mrows * Di * 4);
    unsigned short* ybf   = (unsigned short*)alloc((size_t)Mrows * Di * 2);
    float*          mo    = (float*)alloc((size_t)Mrows * Di * 4);
    (void)ws_size; (void)in_sizes; (void)n_in; (void)out_size;

    auto cvt = [&](const float* src, unsigned short* dst, long long n, long long ntot) {
        k_f32_to_bf16<<<dim3((unsigned)((ntot + 255) / 256)), 256, 0, stream>>>(src, dst, n, ntot);
    };
    cvt(in_proj_w, w_in, (long long)2 * Di * Dd, (long long)2 * Di * Dd);
    cvt(x_proj_w,  w_xp, (long long)(Rr + 2 * Ns) * Di, (long long)128 * Di);
    cvt(dt_proj_w, w_dtp,(long long)Di * Rr, (long long)Di * Rr);
    cvt(out_proj_w,w_op, (long long)Dd * Di, (long long)Dd * Di);
    cvt(fc1_w,     w_f1, (long long)Hh * Dd, (long long)Hh * Dd);
    cvt(fc2_w,     w_f2, (long long)Dd * Hh, (long long)Dd * Hh);

    // 1) gather + LN1
    k_gather_ln<<<Mrows, 256, 0, stream>>>(x, path, ln1);
    // 2) in_proj: xz = ln1 @ in_proj_w^T    (8192 x 2048, K=1024)
    gemm_bf16_wmma<0><<<dim3(2 * Di / 128, Mrows / 128), 256, 0, stream>>>(
        ln1, w_in, xz, nullptr, nullptr, nullptr, Mrows, 2 * Di, Dd);
    // 3) depthwise conv + SiLU
    k_conv_silu<<<dim3((unsigned)(((long long)Mrows * Di + 255) / 256)), 256, 0, stream>>>(
        xz, conv_w, conv_b, xcf, xcb);
    // 4) x_proj: proj = xc @ x_proj_w^T     (8192 x 128(pad), K=1024)
    gemm_bf16_wmma<0><<<dim3(128 / 128, Mrows / 128), 256, 0, stream>>>(
        xcb, w_xp, proj, nullptr, nullptr, nullptr, Mrows, 128, Di);
    // 5) dt_r slice -> bf16
    k_slice_dtr<<<dim3((unsigned)(((long long)Mrows * Rr + 255) / 256)), 256, 0, stream>>>(proj, dtr);
    // 6) dt = softplus(dt_r @ dt_proj_w^T + b)   (8192 x 1024, K=64)
    gemm_bf16_wmma<1><<<dim3(Di / 128, Mrows / 128), 256, 0, stream>>>(
        dtr, w_dtp, dtf, nullptr, dt_proj_b, nullptr, Mrows, Di, Rr);
    // 7) selective scan + gating
    k_scan<<<dim3(Di / 256, Bb), 256, 0, stream>>>(proj, dtf, xcf, xz, A_log, D_param, ybf);
    // 8) out_proj: mo = y @ out_proj_w^T    (8192 x 1024, K=1024)
    gemm_bf16_wmma<0><<<dim3(Dd / 128, Mrows / 128), 256, 0, stream>>>(
        ybf, w_op, mo, nullptr, nullptr, nullptr, Mrows, Dd, Di);
    // 9) residual scatter + LN2 (out holds xres; ln1 buffer reused as ln2)
    k_resid_ln<<<Mrows, 256, 0, stream>>>(x, mo, path_rev, out, ln1);
    // 10) fc1 + GELU -> bf16 (h1 aliases xz buffer)
    unsigned short* h1 = (unsigned short*)xz;
    gemm_bf16_wmma<2><<<dim3(Hh / 128, Mrows / 128), 256, 0, stream>>>(
        ln1, w_f1, nullptr, h1, fc1_b, nullptr, Mrows, Hh, Dd);
    // 11) fc2 + bias + residual -> out
    gemm_bf16_wmma<3><<<dim3(Dd / 128, Mrows / 128), 256, 0, stream>>>(
        h1, w_f2, out, nullptr, fc2_b, out, Mrows, Dd, Hh);
}
